// LilaModel_46299747451031
// MI455X (gfx1250) — compile-verified, matrix operand order checked by reading, not run on hardware
//
#include <hip/hip_runtime.h>

typedef __attribute__((ext_vector_type(16))) _Float16 v16h;
typedef __attribute__((ext_vector_type(8)))  float    v8f;

#define TB    16            // samples per workgroup
#define ROWS  (TB * 36)     // 576 patch-rows per workgroup
#define TILES (ROWS / 16)   // 36 WMMA row tiles

// LDS weight-stash offsets (floats)
#define OFF_W2 0
#define OFF_b2 64
#define OFF_W3 72
#define OFF_b3 80
#define OFF_E1 81
#define OFF_c1 673
#define OFF_E2 689
#define OFF_c2 945
#define OFF_E3 961
#define OFF_c3 977
#define OFF_b1 978
#define NW     986

__device__ __forceinline__ float crelu(float x) {
    return fminf(fmaxf(x, -1.0f), 1.0f);
}

__global__ __launch_bounds__(256) void lila_fused(
    const float* __restrict__ inp,
    const float* __restrict__ W1, const float* __restrict__ b1,
    const float* __restrict__ W2, const float* __restrict__ b2,
    const float* __restrict__ W3, const float* __restrict__ b3,
    const float* __restrict__ E1, const float* __restrict__ c1,
    const float* __restrict__ E2, const float* __restrict__ c2,
    const float* __restrict__ E3, const float* __restrict__ c3,
    float* __restrict__ out)
{
    // f16 staged board features, 4-byte-pair addressable
    __shared__ __attribute__((aligned(16))) _Float16 s_in[TB * 384];
    __shared__ float s_last[TB];       // inp[:,384]
    __shared__ float s_h[ROWS * 8];    // layer-1 activations
    __shared__ float s_v[ROWS];        // per-patch scalar v
    __shared__ float s_g[TB * 16];     // head hidden 1
    __shared__ float s_g2[TB * 16];    // head hidden 2
    __shared__ float s_w[NW];          // small weights

    const int tid  = threadIdx.x;
    const int lane = tid & 31;
    const int wave = tid >> 5;
    const int s0   = blockIdx.x * TB;

    // Warm the input stream (emits global_prefetch_b8)
    __builtin_prefetch(&inp[(size_t)s0 * 385], 0, 1);

    // ---- stage inputs: f32 -> f16 into LDS (coalesced per sample row) ----
    for (int s = 0; s < TB; ++s) {
        const float* row = inp + (size_t)(s0 + s) * 385;
        for (int f = tid; f < 384; f += 256)
            s_in[s * 384 + f] = (_Float16)row[f];
    }
    if (tid < TB) s_last[tid] = inp[(size_t)(s0 + tid) * 385 + 384];

    // ---- stage small weights into LDS ----
    if (tid < 64)  s_w[OFF_W2 + tid] = W2[tid];
    if (tid < 8)   s_w[OFF_b2 + tid] = b2[tid];
    if (tid < 8)   s_w[OFF_W3 + tid] = W3[tid];
    if (tid == 0)  s_w[OFF_b3]       = b3[0];
    for (int i = tid; i < 592; i += 256) s_w[OFF_E1 + i] = E1[i];
    if (tid < 16)  s_w[OFF_c1 + tid] = c1[tid];
    if (tid < 256) s_w[OFF_E2 + tid] = E2[tid];
    if (tid < 16)  s_w[OFF_c2 + tid] = c2[tid];
    if (tid < 16)  s_w[OFF_E3 + tid] = E3[tid];
    if (tid == 0)  s_w[OFF_c3]       = c3[0];
    if (tid < 8)   s_w[OFF_b1 + tid] = b1[tid];

    const int nn = lane & 15;       // N column / tile row index
    const int hi = lane >> 4;       // lane group selects K half

    // ---- build W1 B-fragments (54x8 padded to 64x16), branch-free ----
    union VH { v16h v; _Float16 e[16]; };
    VH bf[2];
#pragma unroll
    for (int kt = 0; kt < 2; ++kt) {
#pragma unroll
        for (int e2 = 0; e2 < 16; ++e2) {
            const int K  = kt * 32 + hi * 8 + (e2 < 8 ? e2 : e2 + 8);
            const int kc = (K < 54) ? K : 0;                 // clamp, load always
            const float w = W1[kc * 8 + nn];
            bf[kt].e[e2] = (K < 54 && nn < 8) ? (_Float16)w : (_Float16)0.0f;
        }
    }

    // ---- loop-invariant pair indices (uint units) for the A gather ----
    // board index of feature K = px*48 + py*6 + K + 30*(K/18): contiguous
    // 18-half runs per ox block; all fragment pairs land on even halves.
    const int i_a0_0 = 4 * hi;               // K = 8*hi + {0..7}
    const int i_a0_4 = hi ? 27 : 8;          // K = 16 + 8*hi
    const int i_a0_5 = 24 + 4 * hi;          // K = 18+8*hi .. 23+8*hi
    const int i_a1_0 = hi ? 50 : 31;         // K1 = 32+8*hi {+0,1}
    const int i_a1_1 = hi ? 51 : 32;         // K1 = 34+8*hi
    const int i_a1_2 = hi ? 52 : 48;         // K1 = 36+8*hi
    const int i_a1_3 = hi ? 53 : 49;         // K1 = 38+8*hi
    const unsigned zmask = hi ? 0u : 0xffffffffu;  // K1 >= 54 tail is hi-only

    const unsigned* __restrict__ s_in32 = (const unsigned*)s_in;

    __syncthreads();

    // ---- layer 1: (576 x 54) x (54 x 8) GEMM via v_wmma_f32_16x16x32_f16 ----
    for (int t = wave; t < TILES; t += 8) {        // wave-uniform trip count
        const int row = t * 16 + nn;               // local patch-row 0..575
        const int smp = row / 36;
        const int p   = row - smp * 36;
        const int px  = p / 6;
        const int py  = p - px * 6;
        // uint index of K=0 for this (sample, patch)
        const unsigned* sp = s_in32 + (smp * 192 + px * 24 + py * 3);

        union VU { v16h v; unsigned u[8]; } a0, a1;
        a0.u[0] = sp[i_a0_0 + 0];
        a0.u[1] = sp[i_a0_0 + 1];
        a0.u[2] = sp[i_a0_0 + 2];
        a0.u[3] = sp[i_a0_0 + 3];
        a0.u[4] = sp[i_a0_4];
        a0.u[5] = sp[i_a0_5 + 0];
        a0.u[6] = sp[i_a0_5 + 1];
        a0.u[7] = sp[i_a0_5 + 2];

        a1.u[0] = sp[i_a1_0];
        a1.u[1] = sp[i_a1_1];
        a1.u[2] = sp[i_a1_2];
        a1.u[3] = sp[i_a1_3];
        a1.u[4] = sp[54] & zmask;                  // K1 = 48,49 (hi=0 only)
        a1.u[5] = sp[55] & zmask;                  // K1 = 50,51
        a1.u[6] = sp[56] & zmask;                  // K1 = 52,53
        a1.u[7] = 0u;                              // K1 = 54..63 pad

        v8f c = {};
        c = __builtin_amdgcn_wmma_f32_16x16x32_f16(false, a0.v, false, bf[0].v,
                                                   (short)0, c, false, false);
        c = __builtin_amdgcn_wmma_f32_16x16x32_f16(false, a1.v, false, bf[1].v,
                                                   (short)0, c, false, false);

        if (nn < 8) {                              // real output columns only
            const float bb = s_w[OFF_b1 + nn];
            union { v8f v; float f[8]; } cu; cu.v = c;
#pragma unroll
            for (int j = 0; j < 8; ++j) {
                const int r = t * 16 + hi * 8 + j;  // C/D: M = j + 8*hi
                s_h[r * 8 + nn] = crelu(cu.f[j] + bb);
            }
        }
    }
    __syncthreads();

    // ---- layers 2 & 3: one (sample,patch) per thread round, VALU f32 ----
    for (int q = tid; q < ROWS; q += 256) {
        float h[8];
#pragma unroll
        for (int i = 0; i < 8; ++i) h[i] = s_h[q * 8 + i];
        float h2[8];
#pragma unroll
        for (int o = 0; o < 8; ++o) {
            float acc = s_w[OFF_b2 + o];
#pragma unroll
            for (int i = 0; i < 8; ++i) acc += h[i] * s_w[OFF_W2 + i * 8 + o];
            h2[o] = crelu(acc);
        }
        float acc = s_w[OFF_b3];
#pragma unroll
        for (int i = 0; i < 8; ++i) acc += h2[i] * s_w[OFF_W3 + i];
        s_v[q] = crelu(acc);
    }
    __syncthreads();

    // ---- head layer 1: 16 samples x 16 neurons = 256 threads ----
    {
        const int smp = tid >> 4;
        const int n   = tid & 15;
        float acc = s_w[OFF_c1 + n];
        const float* vrow = &s_v[smp * 36];
#pragma unroll
        for (int i = 0; i < 36; ++i) acc += vrow[i] * s_w[OFF_E1 + i * 16 + n];
        acc += s_last[smp] * s_w[OFF_E1 + 36 * 16 + n];
        s_g[smp * 16 + n] = crelu(acc);
    }
    __syncthreads();

    // ---- head layer 2 ----
    {
        const int smp = tid >> 4;
        const int n   = tid & 15;
        float acc = s_w[OFF_c2 + n];
#pragma unroll
        for (int i = 0; i < 16; ++i) acc += s_g[smp * 16 + i] * s_w[OFF_E2 + i * 16 + n];
        s_g2[smp * 16 + n] = crelu(acc);
    }
    __syncthreads();

    // ---- head output ----
    if (tid < TB) {
        float acc = s_w[OFF_c3];
#pragma unroll
        for (int i = 0; i < 16; ++i) acc += s_g2[tid * 16 + i] * s_w[OFF_E3 + i];
        out[s0 + tid] = crelu(acc);
    }
}

extern "C" void kernel_launch(void* const* d_in, const int* in_sizes, int n_in,
                              void* d_out, int out_size, void* d_ws, size_t ws_size,
                              hipStream_t stream) {
    const float* inp = (const float*)d_in[0];
    const float* W1  = (const float*)d_in[1];
    const float* b1  = (const float*)d_in[2];
    const float* W2  = (const float*)d_in[3];
    const float* b2  = (const float*)d_in[4];
    const float* W3  = (const float*)d_in[5];
    const float* b3  = (const float*)d_in[6];
    const float* E1  = (const float*)d_in[7];
    const float* c1  = (const float*)d_in[8];
    const float* E2  = (const float*)d_in[9];
    const float* c2  = (const float*)d_in[10];
    const float* E3  = (const float*)d_in[11];
    const float* c3  = (const float*)d_in[12];
    float* out = (float*)d_out;

    const int B = in_sizes[0] / 385;          // 131072
    dim3 grid(B / TB), block(256);
    lila_fused<<<grid, block, 0, stream>>>(inp, W1, b1, W2, b2, W3, b3,
                                           E1, c1, E2, c2, E3, c3, out);
}
